// SemWindowAttention_69432441307687
// MI455X (gfx1250) — matrix operand order repaired and checked by reading, not compiled, not run on hardware
//
#include <hip/hip_runtime.h>

// Problem constants
#define NB    8
#define HH    256
#define WW    256
#define CC    256
#define WS    8
#define NCLS  18
#define NWIN  (NB * (HH / WS) * (WW / WS))   // 8192 windows
#define NTOK  (WS * WS)                      // 64 tokens / window

// LDS layout (bytes): dynamic shared memory, regions aliased by liveness
#define XW_LD 264   // bf16 row stride for xw / Fbf / weight-stage (256 + 8 pad)
#define QK_LD 40    // bf16 row stride for Q/K (32 + 8 pad)
#define SB_LD 72    // bf16 row stride for softmax(S) (64 + 8 pad)
#define SS_LD 68    // f32 row stride for raw scores
#define VT_LD 72    // bf16 row stride for V^T (64 tok + 8 pad)

#define OFF_XW   0                      // [64][264] bf16 = 33792 (xw, later F)
#define OFF_WST  33792                  // [64][264] bf16 = 33792 (B staging)
#define OFF_QK   67584                  // 2 x [64][40] bf16 = 10240
#define OFF_SBF  77824                  // [64][72]  bf16 = 9216
#define OFF_VT   87040                  // [256][72] bf16 = 36864 (overlays f32 S)
#define LDS_BYTES 123904

typedef __attribute__((ext_vector_type(16))) __bf16 v16bf;
typedef __attribute__((ext_vector_type(8)))  float  v8f;

union BF16Vec { v16bf v; uint4 q[2]; };

__device__ __forceinline__ unsigned short f2bf(float f) {
  union { float f; unsigned u; } x; x.f = f;
  unsigned r = x.u + 0x7FFFu + ((x.u >> 16) & 1u);   // round-to-nearest-even
  return (unsigned short)(r >> 16);
}

// A operand (16x32 bf16, M x K). ISA layout: lanes 0-15 row M=lane, halves
// {K=0..7, K=16..23}; lanes 16-31 same rows, halves {K=8..15, K=24..31}.
__device__ __forceinline__ v16bf ldsA(const unsigned short* rowBase, int ld, int k0, int lane) {
  int m = lane & 15;
  int o = (lane & 16) >> 1;                         // 0 or 8
  const unsigned short* p = rowBase + m * ld + k0 + o;
  BF16Vec r;
  r.q[0] = *(const uint4*)(p);                      // K = k0+o .. +7
  r.q[1] = *(const uint4*)(p + 16);                 // K = k0+o+16 .. +23
  return r.v;
}

// B operand (32x16 bf16, K x N), column N contiguous over K in LDS (N-major).
// ISA layout: lane n<16 holds K=0..7,16..23 of col n; lane n+16 holds K=8..15,24..31.
__device__ __forceinline__ v16bf ldsB(const unsigned short* rowBase, int ld, int k0, int lane) {
  int n = lane & 15;
  int kb = k0 + ((lane & 16) >> 1);
  const unsigned short* p = rowBase + n * ld + kb;
  BF16Vec r;
  r.q[0] = *(const uint4*)(p);                      // K = kb .. kb+7
  r.q[1] = *(const uint4*)(p + 16);                 // K = kb+16 .. kb+23
  return r.v;
}

__device__ __forceinline__ v8f wmma_bf16(v16bf a, v16bf b, v8f c) {
  return __builtin_amdgcn_wmma_f32_16x16x32_bf16(false, a, false, b, (short)0, c, false, false);
}

__global__ __launch_bounds__(256)
void semwin_attn_kernel(const float* __restrict__ x,
                        const float* __restrict__ Wq, const float* __restrict__ bq,
                        const float* __restrict__ Wk, const float* __restrict__ bk,
                        const float* __restrict__ Wv, const float* __restrict__ bv,
                        const float* __restrict__ Wres, const float* __restrict__ bres,
                        const float* __restrict__ gamma,
                        float* __restrict__ out_seg, float* __restrict__ out_feat)
{
  extern __shared__ __align__(16) char smem[];
  unsigned short* xwbf = (unsigned short*)(smem + OFF_XW);   // xw bf16, later F bf16
  unsigned short* wst  = (unsigned short*)(smem + OFF_WST);  // transposed weight chunk
  unsigned short* qmat = (unsigned short*)(smem + OFF_QK);   // Q [64][40]
  unsigned short* kmat = qmat + 64 * QK_LD;                  // K [64][40]
  unsigned short* sbf  = (unsigned short*)(smem + OFF_SBF);  // softmax(S) bf16
  float*          ssf  = (float*)(smem + OFF_VT);            // raw S f32 (dies before Vt)
  unsigned short* vt   = (unsigned short*)(smem + OFF_VT);   // V^T [256][72]

  const int tid  = threadIdx.x;
  const int lane = tid & 31;
  const int wave = tid >> 5;

  const int widx = blockIdx.x;
  const int bi  = widx >> 10;
  const int wh  = (widx >> 5) & 31;
  const int wcw = widx & 31;

  const float* xwin = x + (((size_t)bi * HH + wh * WS) * WW + wcw * WS) * CC;

  // ---------- Phase A: load window, convert to bf16 A operand ----------
#pragma unroll
  for (int tt = 0; tt < 8; ++tt) {
    int t = wave * 8 + tt;
    const float* rowp = xwin + ((size_t)(t >> 3) * WW + (t & 7)) * CC;
#pragma unroll
    for (int h = 0; h < 2; ++h) {
      int c4 = lane + h * 32;                       // float4 index in row (0..63)
      float4 v = ((const float4*)rowp)[c4];
      uint2 pk;
      pk.x = (unsigned)f2bf(v.x) | ((unsigned)f2bf(v.y) << 16);
      pk.y = (unsigned)f2bf(v.z) | ((unsigned)f2bf(v.w) << 16);
      *(uint2*)(xwbf + t * XW_LD + c4 * 4) = pk;
    }
  }

  // ---------- Phase B1: stage Wq^T (rows 0..31) and Wk^T (rows 32..63), N padded to 32 ----------
  {
    int k = tid;                                    // 0..255 = input channel
    for (int n = 0; n < 32; ++n) {
      float q = 0.f, kv = 0.f;
      if (n < NCLS) { q = Wq[k * NCLS + n]; kv = Wk[k * NCLS + n]; }
      wst[n * XW_LD + k]        = f2bf(q);
      wst[(32 + n) * XW_LD + k] = f2bf(kv);
    }
  }
  __syncthreads();

  // ---------- Phase B2: Q = xw@Wq+bq, K = xw@Wk+bk ; write seg_map (f32) to global ----------
  for (int t = wave; t < 16; t += 8) {
    int mt = t >> 2, nt = t & 3;                    // nt 0,1 -> Q cols 0..31 ; nt 2,3 -> K
    v8f acc = {0.f,0.f,0.f,0.f,0.f,0.f,0.f,0.f};
#pragma unroll
    for (int kk = 0; kk < 8; ++kk)
      acc = wmma_bf16(ldsA(xwbf + mt * 16 * XW_LD, XW_LD, kk * 32, lane),
                      ldsB(wst  + nt * 16 * XW_LD, XW_LD, kk * 32, lane), acc);
    bool isQ = (nt < 2);
    int col  = (nt & 1) * 16 + (lane & 15);         // 0..31 within Q or K
    float bias = 0.f;
    if (col < NCLS) bias = isQ ? bq[col] : bk[col];
    int m0 = mt * 16 + ((lane & 16) >> 1);
    unsigned short* dmat = isQ ? qmat : kmat;
#pragma unroll
    for (int r = 0; r < 8; ++r) {
      float val = (col < NCLS) ? (acc[r] + bias) : 0.f;  // exact-zero pad cols
      dmat[(m0 + r) * QK_LD + col] = f2bf(val);
      if (isQ && col < NCLS) {
        int m = m0 + r;
        int hh2 = wh * WS + (m >> 3);
        int wc2 = wcw * WS + (m & 7);
        out_seg[(((size_t)bi * HH + hh2) * WW + wc2) * NCLS + col] = val;
      }
    }
  }
  __syncthreads();

  // ---------- Phase B3: S = Q @ K^T (64x64, K-dim 32 incl. zero pad) ----------
  for (int t = wave; t < 16; t += 8) {
    int mt = t >> 2, nt = t & 3;
    v8f acc = {0.f,0.f,0.f,0.f,0.f,0.f,0.f,0.f};
    acc = wmma_bf16(ldsA(qmat + mt * 16 * QK_LD, QK_LD, 0, lane),
                    ldsB(kmat + nt * 16 * QK_LD, QK_LD, 0, lane), acc);
    int n  = nt * 16 + (lane & 15);
    int m0 = mt * 16 + ((lane & 16) >> 1);
#pragma unroll
    for (int r = 0; r < 8; ++r) ssf[(m0 + r) * SS_LD + n] = acc[r];
  }
  __syncthreads();

  // ---------- Phase B4: row softmax -> bf16 A operand ----------
  {
    int row = tid >> 2, seg = tid & 2 ? (tid & 3) : (tid & 3);   // 4 lanes per row
    seg = tid & 3;
    const float* sr = ssf + row * SS_LD + seg * 16;
    float vals[16];
    float mx = -3.402823466e38f;
#pragma unroll
    for (int i = 0; i < 16; ++i) { vals[i] = sr[i]; mx = fmaxf(mx, vals[i]); }
    mx = fmaxf(mx, __shfl_xor(mx, 1));
    mx = fmaxf(mx, __shfl_xor(mx, 2));
    float sum = 0.f;
#pragma unroll
    for (int i = 0; i < 16; ++i) { vals[i] = __expf(vals[i] - mx); sum += vals[i]; }
    sum += __shfl_xor(sum, 1);
    sum += __shfl_xor(sum, 2);
    float inv = 1.f / sum;
    unsigned short* dr = sbf + row * SB_LD + seg * 16;
#pragma unroll
    for (int i = 0; i < 16; i += 2) {
      unsigned pk = (unsigned)f2bf(vals[i] * inv) | ((unsigned)f2bf(vals[i + 1] * inv) << 16);
      *(unsigned*)(dr + i) = pk;
    }
  }
  __syncthreads();

  // ---------- Phase C: V = xw@Wv+bv in 4 column chunks; store V^T for next GEMM ----------
  for (int cn = 0; cn < 4; ++cn) {
    {
      int nloc = tid & 63;
      int kb   = tid >> 6;
      const float* wp = Wv + cn * 64 + nloc;
      for (int k = kb; k < CC; k += 4)
        wst[nloc * XW_LD + k] = f2bf(wp[(size_t)k * CC]);
    }
    __syncthreads();
    for (int t = wave; t < 16; t += 8) {
      int mt = t >> 2, nt = t & 3;
      v8f acc = {0.f,0.f,0.f,0.f,0.f,0.f,0.f,0.f};
#pragma unroll
      for (int kk = 0; kk < 8; ++kk)
        acc = wmma_bf16(ldsA(xwbf + mt * 16 * XW_LD, XW_LD, kk * 32, lane),
                        ldsB(wst  + nt * 16 * XW_LD, XW_LD, kk * 32, lane), acc);
      int c = cn * 64 + nt * 16 + (lane & 15);
      float bias = bv[c];
      int tok0 = mt * 16 + ((lane & 16) >> 1);
      unsigned short h[8];
#pragma unroll
      for (int r = 0; r < 8; ++r) h[r] = f2bf(acc[r] + bias);
      uint4 pk;
      pk.x = (unsigned)h[0] | ((unsigned)h[1] << 16);
      pk.y = (unsigned)h[2] | ((unsigned)h[3] << 16);
      pk.z = (unsigned)h[4] | ((unsigned)h[5] << 16);
      pk.w = (unsigned)h[6] | ((unsigned)h[7] << 16);
      *(uint4*)(vt + c * VT_LD + tok0) = pk;        // transposed: Vt[c][tok0..tok0+7]
    }
    __syncthreads();
  }

  // ---------- Phase D: F = softmax(S) @ V  (64x256, K=64) -> overwrite xwbf ----------
  for (int t = wave; t < 64; t += 8) {
    int mt = t >> 4, nt = t & 15;
    v8f acc = {0.f,0.f,0.f,0.f,0.f,0.f,0.f,0.f};
#pragma unroll
    for (int kk = 0; kk < 2; ++kk)
      acc = wmma_bf16(ldsA(sbf + mt * 16 * SB_LD, SB_LD, kk * 32, lane),
                      ldsB(vt  + nt * 16 * VT_LD, VT_LD, kk * 32, lane), acc);
    int n  = nt * 16 + (lane & 15);
    int m0 = mt * 16 + ((lane & 16) >> 1);
#pragma unroll
    for (int r = 0; r < 8; ++r)
      xwbf[(m0 + r) * XW_LD + n] = f2bf(acc[r]);    // F as next A operand
  }
  __syncthreads();

  // ---------- Phase E: O = F@Wres+bres ; out = gamma*O + x (residual re-read, L2-hot) ----------
  const float gval = gamma[0];
  for (int cn = 0; cn < 4; ++cn) {
    {
      int nloc = tid & 63;
      int kb   = tid >> 6;
      const float* wp = Wres + cn * 64 + nloc;
      for (int k = kb; k < CC; k += 4)
        wst[nloc * XW_LD + k] = f2bf(wp[(size_t)k * CC]);
    }
    __syncthreads();
    for (int t = wave; t < 16; t += 8) {
      int mt = t >> 2, nt = t & 3;
      v8f acc = {0.f,0.f,0.f,0.f,0.f,0.f,0.f,0.f};
#pragma unroll
      for (int kk = 0; kk < 8; ++kk)
        acc = wmma_bf16(ldsA(xwbf + mt * 16 * XW_LD, XW_LD, kk * 32, lane),
                        ldsB(wst  + nt * 16 * XW_LD, XW_LD, kk * 32, lane), acc);
      int c = cn * 64 + nt * 16 + (lane & 15);
      float bias = bres[c];
      int m0 = mt * 16 + ((lane & 16) >> 1);
#pragma unroll
      for (int r = 0; r < 8; ++r) {
        int m = m0 + r;
        int ii = m >> 3, jj = m & 7;
        float xres = xwin[((size_t)ii * WW + jj) * CC + c];
        float oval = gval * (acc[r] + bias) + xres;
        int hh2 = wh * WS + ii;
        int wc2 = wcw * WS + jj;
        out_feat[(((size_t)bi * HH + hh2) * WW + wc2) * CC + c] = oval;
      }
    }
    __syncthreads();
  }
}

extern "C" void kernel_launch(void* const* d_in, const int* in_sizes, int n_in,
                              void* d_out, int out_size, void* d_ws, size_t ws_size,
                              hipStream_t stream) {
  (void)in_sizes; (void)n_in; (void)out_size; (void)d_ws; (void)ws_size;
  const float* x     = (const float*)d_in[0];
  const float* Wq    = (const float*)d_in[1];
  const float* bq    = (const float*)d_in[2];
  const float* Wk    = (const float*)d_in[3];
  const float* bk    = (const float*)d_in[4];
  const float* Wv    = (const float*)d_in[5];
  const float* bv    = (const float*)d_in[6];
  const float* Wres  = (const float*)d_in[7];
  const float* bres  = (const float*)d_in[8];
  const float* gamma = (const float*)d_in[9];

  float* out_seg  = (float*)d_out;
  float* out_feat = out_seg + (size_t)NB * HH * WW * NCLS;

  semwin_attn_kernel<<<dim3(NWIN), dim3(256), LDS_BYTES, stream>>>(
      x, Wq, bq, Wk, bk, Wv, bv, Wres, bres, gamma, out_seg, out_feat);
}
